// KANSSSMBlock_62818191671676
// MI455X (gfx1250) — matrix operand order repaired and checked by reading, not compiled, not run on hardware
//
#include <hip/hip_runtime.h>
#include <hip/hip_bf16.h>
#include <math.h>

// ---------------------------------------------------------------------------
// KAN-SSM block for gfx1250 (MI455X): bf16 WMMA GEMMs for the three KAN
// layers (base+spline folded into one K=in*9 GEMM), async global->LDS
// double-buffered B-tile staging (ASYNCcnt), scalar fp32 selective scan.
// ---------------------------------------------------------------------------

typedef __attribute__((ext_vector_type(16))) __bf16 v16bf;
typedef __attribute__((ext_vector_type(8)))  __bf16 v8bf;
typedef __attribute__((ext_vector_type(8)))  float  v8f;
typedef __attribute__((ext_vector_type(4)))  int    v4i;

// Problem constants (from setup_inputs)
#define BB     2
#define NN     2
#define LL     1024
#define DM     512
#define DI     512
#define DS_    16
#define KK     4
#define DTR    32
#define NCOEF  8          // GS + SO
#define KEXP   9          // 8 spline + 1 silu channel per input feature

#if __has_builtin(__builtin_amdgcn_global_load_async_to_lds_b128)
#define KAN_ASYNC 1
#endif

__device__ __forceinline__ float silu_f(float x) { return x * (1.0f / (1.0f + __expf(-x))); }
__device__ __forceinline__ float softplus_f(float x) { return (x > 20.0f) ? x : log1pf(__expf(x)); }

// 16-byte global -> LDS copy: async (GLOBAL_LOAD_ASYNC_TO_LDS_B128) if the
// toolchain exposes it, otherwise synchronous VGPR round-trip.
// Builtin signature (from clang diagnostic): (v4i AS1*, v4i AS3*, imm, imm).
__device__ __forceinline__ void copy_b128_to_lds(const __bf16* g, __bf16* l) {
#ifdef KAN_ASYNC
  const v4i* gp = (const v4i*)g;   // pointee reinterpret in generic AS
  v4i*       lp = (v4i*)l;
  __builtin_amdgcn_global_load_async_to_lds_b128(
      (__attribute__((address_space(1))) v4i*)gp,
      (__attribute__((address_space(3))) v4i*)lp, 0, 0);
#else
  *(v8bf*)l = *(const v8bf*)g;
#endif
}

template <int N>
__device__ __forceinline__ void wait_async() {
#ifdef KAN_ASYNC
#if __has_builtin(__builtin_amdgcn_s_wait_asynccnt)
  __builtin_amdgcn_s_wait_asynccnt(N);
#else
  asm volatile("s_wait_asynccnt %0" ::"i"(N) : "memory");
#endif
#endif
}

// ---------------------------------------------------------------------------
// Pack KAN weights: W[o, i*9+c] = sw[o,i,c]*sc[o,i] (c<8), bw[o,i] (c==8), bf16
// ---------------------------------------------------------------------------
__global__ void kan_pack_w(const float* __restrict__ bw, const float* __restrict__ sw,
                           const float* __restrict__ sc, __bf16* __restrict__ W, int OI) {
  int idx = blockIdx.x * blockDim.x + threadIdx.x;
  if (idx >= OI) return;
  float s = sc[idx];
  __bf16* o = W + (size_t)idx * KEXP;
#pragma unroll
  for (int c = 0; c < NCOEF; ++c) o[c] = (__bf16)(sw[(size_t)idx * NCOEF + c] * s);
  o[NCOEF] = (__bf16)bw[idx];
}

// ---------------------------------------------------------------------------
// Expand activations: Act[t, f*9+c] = Bspline_c(x) (c<8), silu(x) (c==8)
// Uniform grid g[j] = 0.4*(j-3) - 1, Cox-de Boor to order 3 -> 8 bases.
// ---------------------------------------------------------------------------
__global__ void kan_expand(const float* __restrict__ X, __bf16* __restrict__ Act, int total) {
  int idx = blockIdx.x * blockDim.x + threadIdx.x;
  if (idx >= total) return;
  float x = X[idx];
  float b[11];
#pragma unroll
  for (int j = 0; j < 11; ++j) {
    float gj  = 0.4f * (float)(j - 3) - 1.0f;
    float gj1 = 0.4f * (float)(j - 2) - 1.0f;
    b[j] = (x >= gj && x < gj1) ? 1.0f : 0.0f;
  }
#pragma unroll
  for (int k = 1; k <= 3; ++k) {
    float inv = 1.0f / (0.4f * (float)k);
#pragma unroll
    for (int j = 0; j < 10; ++j) {
      if (j < 11 - k) {
        float gj   = 0.4f * (float)(j - 3) - 1.0f;       // g[j]
        float gjk1 = 0.4f * (float)(j + k - 2) - 1.0f;   // g[j+k+1]
        b[j] = (x - gj) * inv * b[j] + (gjk1 - x) * inv * b[j + 1];
      }
    }
  }
  __bf16* o = Act + (size_t)idx * KEXP;
#pragma unroll
  for (int c = 0; c < NCOEF; ++c) o[c] = (__bf16)b[c];
  o[NCOEF] = (__bf16)silu_f(x);
}

// ---------------------------------------------------------------------------
// bf16 WMMA GEMM: C[M,N] = A[M,K] * W[N,K]^T, f32 accumulate.
// Block = 8 waves (4x2) -> 128x64 C tile; wave -> 2x2 wmma 16x16 tiles.
// B tile double-buffered in LDS via async global->LDS copies; the next K-tile
// copy is issued before computing the current one and drained with
// s_wait_asynccnt 1, so the fetch overlaps A loads + 4 WMMAs.
// Requires M%128==0, N%64==0, K%32==0 (true for all three KAN GEMMs).
// ---------------------------------------------------------------------------
#define LDSB_STRIDE 40   // 64 rows x 32 k, padded: 80B row pitch kills bank conflicts

__global__ __launch_bounds__(256) void gemm_kan(const __bf16* __restrict__ A,
                                                const __bf16* __restrict__ W,
                                                float* __restrict__ C,
                                                int M, int N, int K) {
  __shared__ __bf16 ldsB[2][64 * LDSB_STRIDE];
  const int tid  = threadIdx.x;
  const int lane = tid & 31;
  const int wave = tid >> 5;
  const int wm = wave & 3, wn = wave >> 2;
  const int m0      = blockIdx.y * 128 + wm * 32;
  const int n0block = blockIdx.x * 64;
  const int n0      = n0block + wn * 32;

  const int half  = lane >> 4;     // which half-wave
  const int l15   = lane & 15;
  const int kbA   = half * 8;      // A frag K base (lanes16-31 start at K=8)
  const int koffB = half * 8;      // B frag K base (lanes16-31 start at K=8)

  // per-thread slice of the cooperative 64x32 B-tile copy (16 bytes each)
  const int nrow = tid >> 2;           // 0..63
  const int kc   = (tid & 3) * 8;      // 0,8,16,24
  const __bf16* wsrc = W + (size_t)(n0block + nrow) * K + kc;
  __bf16* ldst0 = &ldsB[0][nrow * LDSB_STRIDE + kc];
  __bf16* ldst1 = &ldsB[1][nrow * LDSB_STRIDE + kc];

  v8f acc[2][2] = {};

  copy_b128_to_lds(wsrc, ldst0);       // stage K-tile 0 into buffer 0

  int buf = 0;
  for (int k0 = 0; k0 < K; k0 += 32, buf ^= 1) {
    if (k0 + 32 < K) {
      // issue next tile into the other buffer (safe: its readers finished at
      // the previous iteration's trailing barrier), then wait only for the
      // current tile, leaving the new copy in flight behind the compute.
      copy_b128_to_lds(wsrc + k0 + 32, buf ? ldst0 : ldst1);
      wait_async<1>();
      __builtin_prefetch(A + (size_t)(m0 + l15) * K + k0 + 32, 0, 1);
    } else {
      wait_async<0>();
    }
    __syncthreads();   // current-tile LDS contents visible to all waves

    const __bf16* ldsCur = &ldsB[buf][0];
    v16bf afrag[2];
#pragma unroll
    for (int ti = 0; ti < 2; ++ti) {
      const __bf16* arow = A + (size_t)(m0 + ti * 16 + l15) * K + k0 + kbA;
      v8bf lo = *(const v8bf*)(arow);        // K = kbA .. kbA+7
      v8bf hi = *(const v8bf*)(arow + 16);   // K = kbA+16 .. kbA+23
      afrag[ti] = __builtin_shufflevector(lo, hi, 0, 1, 2, 3, 4, 5, 6, 7,
                                          8, 9, 10, 11, 12, 13, 14, 15);
    }
    v16bf bfrag[2];
#pragma unroll
    for (int tj = 0; tj < 2; ++tj) {
      const __bf16* brow = ldsCur + (wn * 32 + tj * 16 + l15) * LDSB_STRIDE + koffB;
      v8bf lo = *(const v8bf*)(brow);
      v8bf hi = *(const v8bf*)(brow + 16);
      bfrag[tj] = __builtin_shufflevector(lo, hi, 0, 1, 2, 3, 4, 5, 6, 7,
                                          8, 9, 10, 11, 12, 13, 14, 15);
    }
#pragma unroll
    for (int ti = 0; ti < 2; ++ti)
#pragma unroll
      for (int tj = 0; tj < 2; ++tj)
        acc[ti][tj] = __builtin_amdgcn_wmma_f32_16x16x32_bf16(
            false, afrag[ti], false, bfrag[tj], (short)0, acc[ti][tj], false, false);

    __syncthreads();   // all reads of 'buf' done before it is rewritten
  }

  // C/D layout: vgpr r, lanes0-15 -> (M=r, N=lane); lanes16-31 -> (M=r+8, N=lane-16)
#pragma unroll
  for (int ti = 0; ti < 2; ++ti)
#pragma unroll
    for (int tj = 0; tj < 2; ++tj)
#pragma unroll
      for (int r = 0; r < 8; ++r) {
        int m = m0 + ti * 16 + r + half * 8;
        int n = n0 + tj * 16 + l15;
        C[(size_t)m * N + n] = acc[ti][tj][r];
      }
}

// ---------------------------------------------------------------------------
// Causal depthwise conv1d (d_conv=4) + SiLU; writes xs = [fwd ; L-flipped]
// ---------------------------------------------------------------------------
__global__ void conv_silu_xs(const float* __restrict__ xz, const float* __restrict__ cw,
                             const float* __restrict__ cb, float* __restrict__ xs) {
  int idx = blockIdx.x * blockDim.x + threadIdx.x;   // over 4*1024*512 = 2^21
  if (idx >= BB * NN * LL * DI) return;
  int d  = idx & (DI - 1);
  int l  = (idx >> 9) & (LL - 1);
  int bn = idx >> 19;                                // b*2+n
  float acc = cb[d];
#pragma unroll
  for (int j = 0; j < 4; ++j) {
    int ll = l - 3 + j;
    if (ll >= 0) acc += cw[d * 4 + j] * xz[((size_t)(bn * LL + ll)) * (2 * DI) + d];
  }
  float v = silu_f(acc);
  int b = bn >> 1, n = bn & 1;
  xs[((size_t)((b * KK + n)     * LL + l)           ) * DI + d] = v;  // forward scans k=0,1
  xs[((size_t)((b * KK + n + 2) * LL + (LL - 1 - l))) * DI + d] = v;  // reversed  k=2,3
}

// ---------------------------------------------------------------------------
// delta = softplus(dt @ dt_w^T + dt_bias); dt = x_dbl[:, :32] (K=32 VALU dot)
// ---------------------------------------------------------------------------
__global__ void delta_kernel(const float* __restrict__ x_dbl, const float* __restrict__ dt_w,
                             const float* __restrict__ dt_bias, float* __restrict__ delta) {
  int idx = blockIdx.x * blockDim.x + threadIdx.x;   // over 8192*512 = 2^22
  if (idx >= BB * KK * LL * DI) return;
  int d = idx & (DI - 1);
  int t = idx >> 9;
  int k = (t >> 10) & 3;
  float s = dt_bias[k * DI + d];
  const float* xr = x_dbl + (size_t)t * 64;
  const float* wr = dt_w + (size_t)d * DTR;
#pragma unroll
  for (int r = 0; r < DTR; ++r) s += xr[r] * wr[r];
  delta[idx] = softplus_f(s);
}

// ---------------------------------------------------------------------------
// Selective scan: one thread per (b,k,d) channel, 16-state recurrence in regs.
// ---------------------------------------------------------------------------
__global__ void scan_kernel(const float* __restrict__ xs, const float* __restrict__ delta,
                            const float* __restrict__ x_dbl, const float* __restrict__ A_logs,
                            float* __restrict__ ys) {
  int idx = blockIdx.x * blockDim.x + threadIdx.x;   // 4096 channels
  if (idx >= BB * KK * DI) return;
  int d = idx & (DI - 1);
  int bk = idx >> 9;                                 // b*4+k
  int k = bk & 3;
  float A[DS_], h[DS_];
#pragma unroll
  for (int s = 0; s < DS_; ++s) {
    A[s] = -__expf(A_logs[(size_t)(k * DI + d) * DS_ + s]);
    h[s] = 0.0f;
  }
  const size_t t0 = (size_t)bk * LL;
  for (int l = 0; l < LL; ++l) {
    size_t t = t0 + l;
    float dl = delta[t * DI + d];
    float u  = xs[t * DI + d];
    float du = dl * u;
    const float* Brow = x_dbl + t * 64 + DTR;        // cols 32..47
    const float* Crow = x_dbl + t * 64 + DTR + DS_;  // cols 48..63
    float y = 0.0f;
#pragma unroll
    for (int s = 0; s < DS_; ++s) {
      h[s] = h[s] * __expf(dl * A[s]) + du * Brow[s];
      y += h[s] * Crow[s];
    }
    ys[t * DI + d] = y;
  }
}

// ---------------------------------------------------------------------------
// Merge fwd/bwd scans + skip (Ds) and concat z -> out_cat (B,2,L,1024)
// ---------------------------------------------------------------------------
__global__ void merge_kernel(const float* __restrict__ ys, const float* __restrict__ xs,
                             const float* __restrict__ Ds, const float* __restrict__ xz,
                             float* __restrict__ out_cat) {
  int idx = blockIdx.x * blockDim.x + threadIdx.x;   // over 2^21
  if (idx >= BB * NN * LL * DI) return;
  int d  = idx & (DI - 1);
  int l  = (idx >> 9) & (LL - 1);
  int bn = idx >> 19;
  int b = bn >> 1, n = bn & 1;
  size_t tf = (size_t)(b * KK + n) * LL + l;
  size_t tb = (size_t)(b * KK + n + 2) * LL + (LL - 1 - l);
  float yv = (ys[tf * DI + d] + xs[tf * DI + d] * Ds[n * DI + d]) +
             (ys[tb * DI + d] + xs[tb * DI + d] * Ds[(n + 2) * DI + d]);
  size_t to = (size_t)(bn * LL + l);
  out_cat[to * (2 * DI) + d]      = yv;
  out_cat[to * (2 * DI) + DI + d] = xz[to * (2 * DI) + DI + d];   // z
}

// ---------------------------------------------------------------------------
extern "C" void kernel_launch(void* const* d_in, const int* in_sizes, int n_in,
                              void* d_out, int out_size, void* d_ws, size_t ws_size,
                              hipStream_t stream) {
  (void)in_sizes; (void)n_in; (void)out_size; (void)ws_size;
  const float* hidden = (const float*)d_in[0];
  const float* in_bw  = (const float*)d_in[1];
  const float* in_sw  = (const float*)d_in[2];
  const float* in_sc  = (const float*)d_in[3];
  const float* conv_w = (const float*)d_in[4];
  const float* conv_b = (const float*)d_in[5];
  const float* x_bw   = (const float*)d_in[6];
  const float* x_sw   = (const float*)d_in[7];
  const float* x_sc   = (const float*)d_in[8];
  const float* dt_w   = (const float*)d_in[9];
  const float* dt_bias= (const float*)d_in[10];
  const float* A_logs = (const float*)d_in[11];
  const float* Ds     = (const float*)d_in[12];
  const float* out_bw = (const float*)d_in[13];
  const float* out_sw = (const float*)d_in[14];
  const float* out_sc = (const float*)d_in[15];
  float* out = (float*)d_out;

  const int T1 = BB * NN * LL;          // 4096 tokens (in / out KAN)
  const int T2 = BB * KK * LL;          // 8192 tokens (x KAN)
  const int KIN  = DM * KEXP;           // 4608
  const int KX   = DI * KEXP;           // 4608
  const int KOUT = 2 * DI * KEXP;       // 9216

  // workspace arena (256B aligned slots)
  char* ws = (char*)d_ws;
  size_t off = 0;
  auto alloc = [&](size_t bytes) { void* p = ws + off; off += (bytes + 255) & ~(size_t)255; return p; };
  __bf16* W_in  = (__bf16*)alloc((size_t)(2 * DI) * KIN * 2);        //  9.4 MB
  __bf16* W_x   = (__bf16*)alloc((size_t)(DTR + 2 * DS_) * KX * 2);  //  0.6 MB
  __bf16* W_out = (__bf16*)alloc((size_t)DM * KOUT * 2);             //  9.4 MB
  __bf16* Act   = (__bf16*)alloc((size_t)T2 * KX * 2);               // 75.5 MB (reused x3)
  float*  xz    = (float*)alloc((size_t)T1 * 2 * DI * 4);            // 16.8 MB
  float*  xs    = (float*)alloc((size_t)T2 * DI * 4);                // 16.8 MB
  float*  x_dbl = (float*)alloc((size_t)T2 * 64 * 4);                //  2.1 MB
  float*  delta = (float*)alloc((size_t)T2 * DI * 4);                // 16.8 MB
  float*  ys    = (float*)alloc((size_t)T2 * DI * 4);                // 16.8 MB
  float*  ocat  = (float*)alloc((size_t)T1 * 2 * DI * 4);            // 16.8 MB

  const int TPB = 256;
  // 1. pack weights (bf16, base+spline interleaved)
  kan_pack_w<<<(2 * DI * DM + TPB - 1) / TPB, TPB, 0, stream>>>(in_bw, in_sw, in_sc, W_in, 2 * DI * DM);
  kan_pack_w<<<((DTR + 2 * DS_) * DI + TPB - 1) / TPB, TPB, 0, stream>>>(x_bw, x_sw, x_sc, W_x, (DTR + 2 * DS_) * DI);
  kan_pack_w<<<(DM * 2 * DI + TPB - 1) / TPB, TPB, 0, stream>>>(out_bw, out_sw, out_sc, W_out, DM * 2 * DI);

  // 2. KAN-in: expand + GEMM -> xz (4096 x 1024)
  kan_expand<<<(T1 * DM + TPB - 1) / TPB, TPB, 0, stream>>>(hidden, Act, T1 * DM);
  gemm_kan<<<dim3((2 * DI) / 64, T1 / 128), TPB, 0, stream>>>(Act, W_in, xz, T1, 2 * DI, KIN);

  // 3. causal depthwise conv + silu -> xs (fwd + flipped)
  conv_silu_xs<<<(BB * NN * LL * DI + TPB - 1) / TPB, TPB, 0, stream>>>(xz, conv_w, conv_b, xs);

  // 4. KAN-x: expand + GEMM -> x_dbl (8192 x 64)
  kan_expand<<<(T2 * DI + TPB - 1) / TPB, TPB, 0, stream>>>(xs, Act, T2 * DI);
  gemm_kan<<<dim3(64 / 64, T2 / 128), TPB, 0, stream>>>(Act, W_x, x_dbl, T2, 64, KX);

  // 5. delta = softplus(dt @ dt_w^T + bias)
  delta_kernel<<<(T2 * DI + TPB - 1) / TPB, TPB, 0, stream>>>(x_dbl, dt_w, dt_bias, delta);

  // 6. selective scan (serial over L, parallel over 4096 channels)
  scan_kernel<<<(BB * KK * DI + TPB - 1) / TPB, TPB, 0, stream>>>(xs, delta, x_dbl, A_logs, ys);

  // 7. merge scans + concat z
  merge_kernel<<<(BB * NN * LL * DI + TPB - 1) / TPB, TPB, 0, stream>>>(ys, xs, Ds, xz, ocat);

  // 8. KAN-out: expand + GEMM -> d_out (4096 x 512)
  kan_expand<<<(T1 * 2 * DI + TPB - 1) / TPB, TPB, 0, stream>>>(ocat, Act, T1 * 2 * DI);
  gemm_kan<<<dim3(DM / 64, T1 / 128), TPB, 0, stream>>>(Act, W_out, out, T1, DM, KOUT);
}
